// Q_Network_2164663517984
// MI455X (gfx1250) — matrix operand order repaired and checked by reading, not compiled, not run on hardware
//
#include <hip/hip_runtime.h>
#include <cstdint>
#include <cstddef>

// ---------------------------------------------------------------------------
// Q-Network forward for MI455X (gfx1250, wave32, WMMA + TDM).
// bf16 WMMA (16x16x32, f32 accum) for every GEMM. Activation row-blocks are
// DMA'd into LDS by the Tensor Data Mover (one descriptor per WG), weights
// stream from global (L2-resident, ~13 MB). kwta / argmax epilogues fused
// into the GEMM workgroup (each WG owns 16 full rows).
// ---------------------------------------------------------------------------

#define BROWS 16384
#define SDIM  128

typedef __attribute__((ext_vector_type(16))) __bf16        v16bf;
typedef __attribute__((ext_vector_type(8)))  float         v8f;
typedef __attribute__((ext_vector_type(4)))  unsigned int  v4u;
typedef __attribute__((ext_vector_type(4)))  unsigned int  u32x4;
typedef __attribute__((ext_vector_type(8)))  int           i32x8;
typedef __attribute__((ext_vector_type(4)))  int           i32x4;

union ABf { v4u u[2]; v16bf bf; };

// Monotone float -> uint key: descending float order == descending uint order.
__device__ __forceinline__ unsigned fkey(float v) {
    unsigned u = __float_as_uint(v);
    return (u & 0x80000000u) ? ~u : (u | 0x80000000u);
}

// Cheap hash noise in [-0.01, 0.01) (stand-in for the reference's PRNG).
__device__ __forceinline__ float noise_val(unsigned row, unsigned col, unsigned seed) {
    unsigned h = (row * 2654435761u) ^ ((col + 1u) * 2246822519u) ^ (seed * 3266489917u);
    h ^= h >> 13; h *= 0x9E3779B1u; h ^= h >> 16;
    return (((h & 0xFFFFFFu) * (1.0f / 8388608.0f)) - 1.0f) * 0.01f;
}

// ---------------------------------------------------------------------------
// Tensor Data Mover: 2D tile (rows x rowElems bf16) global -> LDS.
// D# layout per CDNA5 ISA (group0/group1 bitfields).
// Tracked by TENSORcnt. EXEC-independent, one issue per wave.
// ---------------------------------------------------------------------------
__device__ __forceinline__ void tdm_load_2d(const void* gaddr, unsigned ldsByteOff,
                                            unsigned rowElems, unsigned rows,
                                            unsigned strideElems, unsigned tensorRows) {
    unsigned long long ga = (unsigned long long)gaddr;
    u32x4 g0;
    g0[0] = 1u;                                         // count=1, user descriptor
    g0[1] = ldsByteOff;                                 // lds_addr [63:32]
    g0[2] = (unsigned)(ga & 0xFFFFFFFFu);               // global_addr lo
    g0[3] = (unsigned)((ga >> 32) & 0x1FFFFFFu) | (2u << 30);   // addr hi | type=2
    i32x8 g1;
    g1[0] = (int)(1u << 16);                            // data_size=1 (2 bytes)
    g1[1] = (int)((rowElems & 0xFFFFu) << 16);          // tensor_dim0[15:0]
    g1[2] = (int)((rowElems >> 16) | ((tensorRows & 0xFFFFu) << 16));  // td0 hi | td1 lo
    g1[3] = (int)((tensorRows >> 16) | ((rowElems & 0xFFFFu) << 16));  // td1 hi | tile_dim0
    g1[4] = (int)(rows & 0xFFFFu);                      // tile_dim1 (tile_dim2=0)
    g1[5] = (int)strideElems;                           // tensor_dim0_stride lo32
    g1[6] = 0;
    g1[7] = 0;
    i32x4 z4 = (i32x4)0;
#if __clang_major__ >= 23
    __builtin_amdgcn_tensor_load_to_lds(g0, g1, z4, z4, (i32x8)0, 0);
#else
    __builtin_amdgcn_tensor_load_to_lds(g0, g1, z4, z4, 0);
#endif
}

// A-tile (16x32 bf16) read from the LDS-staged row block [16, K].
// CDNA5 16-bit A layout: lo lanes rows 0-15 K{0..7,16..23}, hi lanes K{8..15,24..31}
// -> two ds_load_b128 per lane.
__device__ __forceinline__ ABf load_a_lds(const __bf16* aS, int ldk, int k0, int lane) {
    const int r  = lane & 15;
    const int hi = lane >> 4;
    const __bf16* p = aS + r * ldk + k0 + hi * 8;
    ABf a;
    a.u[0] = *reinterpret_cast<const v4u*>(p);
    a.u[1] = *reinterpret_cast<const v4u*>(p + 16);
    return a;
}

// B-tile (32x16 bf16, K x N) with B[k,n] = W[n,k], W row-major [N, ldk].
// Lanes 0-15: column n=lane, K{0..7,16..23}; lanes 16-31: K{8..15,24..31} (matches A).
__device__ __forceinline__ ABf load_b_tile(const __bf16* W, int ldk, int n0, int k0, int lane) {
    const int c  = lane & 15;
    const int hi = lane >> 4;
    const __bf16* p = W + (size_t)(n0 + c) * ldk + k0 + hi * 8;
    ABf b;
    b.u[0] = *reinterpret_cast<const v4u*>(p);
    b.u[1] = *reinterpret_cast<const v4u*>(p + 16);
    return b;
}

// ---------------------------------------------------------------------------
// f32 -> bf16 conversion (weights + state, once per call)
// ---------------------------------------------------------------------------
__global__ void cvt_bf16_kernel(const float* __restrict__ s, __bf16* __restrict__ d, long n) {
    long i = (long)blockIdx.x * blockDim.x + threadIdx.x;
    if (i < n) d[i] = (__bf16)s[i];
}

// ---------------------------------------------------------------------------
// Context nets: first layer (fin=4) + tanh is trivial VALU work
// ---------------------------------------------------------------------------
__global__ void ctx_hidden_kernel(const float* __restrict__ task, const float* __restrict__ W1,
                                  const float* __restrict__ b1, __bf16* __restrict__ h, int width) {
    long idx = (long)blockIdx.x * blockDim.x + threadIdx.x;
    if (idx >= (long)BROWS * width) return;
    int row = (int)(idx / width);
    int n   = (int)(idx % width);
    const float* t = task + (size_t)row * 4;
    const float* w = W1 + (size_t)n * 4;
    float s = b1[n] + t[0] * w[0] + t[1] * w[1] + t[2] * w[2] + t[3] * w[3];
    h[idx] = (__bf16)tanhf(s);
}

// cx4 is 4->32->32: one thread per row, pure VALU.
__global__ void ctx4_kernel(const float* __restrict__ task,
                            const float* __restrict__ W1, const float* __restrict__ b1,
                            const float* __restrict__ W2, const float* __restrict__ b2,
                            int* __restrict__ kOut) {
    int row = blockIdx.x * blockDim.x + threadIdx.x;
    if (row >= BROWS) return;
    float t0 = task[row * 4 + 0], t1 = task[row * 4 + 1];
    float t2 = task[row * 4 + 2], t3 = task[row * 4 + 3];
    float h[32];
    #pragma unroll
    for (int n = 0; n < 32; ++n)
        h[n] = tanhf(b1[n] + t0 * W1[n * 4 + 0] + t1 * W1[n * 4 + 1]
                            + t2 * W1[n * 4 + 2] + t3 * W1[n * 4 + 3]);
    float best = -3.4e38f; int bi = 0;
    for (int n = 0; n < 32; ++n) {
        float y = b2[n];
        #pragma unroll
        for (int j = 0; j < 32; ++j) y += h[j] * W2[n * 32 + j];
        if (y > best) { best = y; bi = n; }
    }
    kOut[row] = bi;
}

// ---------------------------------------------------------------------------
// Context second GEMM (WIDTH x WIDTH) with fused per-row argmax epilogue.
// Block = 256 (8 waves); WG owns 16 full rows. A row-block staged by TDM.
// Shared memory is reused: [bf16 A 16xK] during the loop, [f32 16xN] epilogue.
// ---------------------------------------------------------------------------
template<int N, int K, int NP>
__launch_bounds__(256)
__global__ void gemm_argmax_kernel(const __bf16* __restrict__ A, const __bf16* __restrict__ W,
                                   const float* __restrict__ bias, int* __restrict__ kOut) {
    extern __shared__ char smem_raw[];
    __bf16* aS = (__bf16*)smem_raw;
    float*  fS = (float*)smem_raw;
    const int lane  = threadIdx.x & 31;
    const int wid   = threadIdx.x >> 5;
    const int row0  = blockIdx.x * 16;
    const int nbase = wid * (NP * 16);

    if (wid == 0) {
        tdm_load_2d(A + (size_t)row0 * K, __builtin_amdgcn_groupstaticsize(),
                    K, 16, K, BROWS);
        __builtin_amdgcn_s_wait_tensorcnt(0);
    }
    __syncthreads();

    v8f acc[NP] = {};
    #pragma unroll 2
    for (int k0 = 0; k0 < K; k0 += 32) {
        ABf a = load_a_lds(aS, K, k0, lane);
        #pragma unroll
        for (int t = 0; t < NP; ++t) {
            ABf b = load_b_tile(W, K, nbase + t * 16, k0, lane);
            acc[t] = __builtin_amdgcn_wmma_f32_16x16x32_bf16(false, a.bf, false, b.bf,
                                                             (short)0, acc[t], false, false);
        }
    }
    __syncthreads();                                 // A region -> f32 epilogue reuse
    const int rbase = (lane >> 4) * 8;               // acc lanes 16-31 hold rows 8-15
    #pragma unroll
    for (int t = 0; t < NP; ++t) {
        int n = nbase + t * 16 + (lane & 15);
        float bv = bias[n];
        #pragma unroll
        for (int r = 0; r < 8; ++r) fS[(rbase + r) * N + n] = acc[t][r] + bv;
    }
    __syncthreads();

    #pragma unroll
    for (int rr = 0; rr < 2; ++rr) {                 // 2 rows per wave
        const int row  = wid * 2 + rr;
        const float* rowp = fS + row * N;
        float best = -3.4e38f; int bi = 0;
        for (int idx = lane; idx < N; idx += 32) {
            float v = rowp[idx];
            if (v > best || (v == best && idx < bi)) { best = v; bi = idx; }
        }
        #pragma unroll
        for (int off = 16; off > 0; off >>= 1) {
            float ov = __shfl_xor(best, off, 32);
            int   oi = __shfl_xor(bi, off, 32);
            if (ov > best || (ov == best && oi < bi)) { best = ov; bi = oi; }
        }
        if (lane == 0) kOut[row0 + row] = bi;
    }
}

// ---------------------------------------------------------------------------
// Main-chain layer: GEMM + bias + kwta (dynamic per-row k) + bf16 store,
// optionally appending the bf16 state tail (the concat in the reference).
// 32-bit radix select over monotone keys, wave-shuffle reductions.
// ---------------------------------------------------------------------------
template<int N, int K, int NP, bool TAIL>
__launch_bounds__(256)
__global__ void gemm_kwta_kernel(const __bf16* __restrict__ A, const __bf16* __restrict__ W,
                                 const float* __restrict__ bias, const int* __restrict__ kArr,
                                 const __bf16* __restrict__ stateBF,
                                 __bf16* __restrict__ Out, int ldo, unsigned seed) {
    extern __shared__ char smem_raw[];
    __bf16* aS = (__bf16*)smem_raw;
    float*  fS = (float*)smem_raw;
    const int lane  = threadIdx.x & 31;
    const int wid   = threadIdx.x >> 5;
    const int row0  = blockIdx.x * 16;
    const int nbase = wid * (NP * 16);

    if (wid == 0) {
        tdm_load_2d(A + (size_t)row0 * K, __builtin_amdgcn_groupstaticsize(),
                    K, 16, K, BROWS);
        __builtin_amdgcn_s_wait_tensorcnt(0);
    }
    __syncthreads();

    v8f acc[NP] = {};
    #pragma unroll 2
    for (int k0 = 0; k0 < K; k0 += 32) {
        ABf a = load_a_lds(aS, K, k0, lane);
        #pragma unroll
        for (int t = 0; t < NP; ++t) {
            ABf b = load_b_tile(W, K, nbase + t * 16, k0, lane);
            acc[t] = __builtin_amdgcn_wmma_f32_16x16x32_bf16(false, a.bf, false, b.bf,
                                                             (short)0, acc[t], false, false);
        }
    }
    __syncthreads();                                 // A region -> f32 epilogue reuse
    const int rbase = (lane >> 4) * 8;
    #pragma unroll
    for (int t = 0; t < NP; ++t) {
        int n = nbase + t * 16 + (lane & 15);
        float bv = bias[n];
        #pragma unroll
        for (int r = 0; r < 8; ++r) fS[(rbase + r) * N + n] = acc[t][r] + bv;
    }
    __syncthreads();

    #pragma unroll
    for (int rr = 0; rr < 2; ++rr) {                 // 2 rows per wave
        const int row  = wid * 2 + rr;
        const int grow = row0 + row;
        const int kk   = kArr[grow];
        const float* rowp = fS + row * N;

        // Per-lane key cache, then 32-pass radix select for the largest
        // threshold T with |{key >= T}| >= kk  (T == kk-th largest key).
        unsigned keys[N / 32];
        #pragma unroll
        for (int i = 0; i < N / 32; ++i) keys[i] = fkey(rowp[lane + 32 * i]);

        unsigned T = 0u;
        if (kk > 0) {
            for (int bit = 31; bit >= 0; --bit) {
                unsigned cand = T | (1u << bit);
                int cnt = 0;
                #pragma unroll
                for (int i = 0; i < N / 32; ++i) cnt += (keys[i] >= cand) ? 1 : 0;
                #pragma unroll
                for (int off = 16; off > 0; off >>= 1) cnt += __shfl_xor(cnt, off, 32);
                if (cnt >= kk) T = cand;
            }
        }
        #pragma unroll
        for (int i = 0; i < N / 32; ++i) {
            int idx = lane + 32 * i;
            float v = rowp[idx];
            bool keep = (kk > 0) && (keys[i] >= T);
            float o = keep ? v : noise_val((unsigned)grow, (unsigned)idx, seed);
            Out[(size_t)grow * ldo + idx] = (__bf16)o;
        }
        if (TAIL) {                                  // concat([x, state]) tail
            for (int j = lane; j < SDIM; j += 32)
                Out[(size_t)grow * ldo + N + j] = stateBF[(size_t)grow * SDIM + j];
        }
    }
}

// ---------------------------------------------------------------------------
// linear5: [B,512] -> [B,32] f32 output. 8 waves x 16 rows, 2 column tiles.
// Per-wave TDM staging (each wave owns a private 16-row block + LDS slice);
// no barrier needed, just s_wait_tensorcnt.
// ---------------------------------------------------------------------------
__launch_bounds__(256)
__global__ void gemm_out_kernel(const __bf16* __restrict__ A, const __bf16* __restrict__ W,
                                const float* __restrict__ bias, float* __restrict__ out) {
    extern __shared__ char smem_raw[];
    const int lane = threadIdx.x & 31;
    const int wid  = threadIdx.x >> 5;
    const int row0 = blockIdx.x * 128 + wid * 16;
    __bf16* aS = (__bf16*)smem_raw + (size_t)wid * 16 * 512;

    tdm_load_2d(A + (size_t)row0 * 512,
                __builtin_amdgcn_groupstaticsize() + (unsigned)wid * 16 * 512 * 2,
                512, 16, 512, BROWS);
    __builtin_amdgcn_s_wait_tensorcnt(0);

    v8f acc[2] = {};
    #pragma unroll 2
    for (int k0 = 0; k0 < 512; k0 += 32) {
        ABf a = load_a_lds(aS, 512, k0, lane);
        #pragma unroll
        for (int t = 0; t < 2; ++t) {
            ABf b = load_b_tile(W, 512, t * 16, k0, lane);
            acc[t] = __builtin_amdgcn_wmma_f32_16x16x32_bf16(false, a.bf, false, b.bf,
                                                             (short)0, acc[t], false, false);
        }
    }
    const int rbase = (lane >> 4) * 8;
    #pragma unroll
    for (int t = 0; t < 2; ++t) {
        int n = t * 16 + (lane & 15);
        float bv = bias[n];
        #pragma unroll
        for (int r = 0; r < 8; ++r)
            out[(size_t)(row0 + rbase + r) * 32 + n] = acc[t][r] + bv;
    }
}

// ---------------------------------------------------------------------------
// Host-side orchestration
// ---------------------------------------------------------------------------
extern "C" void kernel_launch(void* const* d_in, const int* in_sizes, int n_in,
                              void* d_out, int out_size, void* d_ws, size_t ws_size,
                              hipStream_t stream) {
    (void)in_sizes; (void)n_in; (void)out_size; (void)ws_size;
    const float* state  = (const float*)d_in[0];
    const float* task   = (const float*)d_in[1];
    const float* cx1_1w = (const float*)d_in[2],  *cx1_1b = (const float*)d_in[3];
    const float* cx1_2w = (const float*)d_in[4],  *cx1_2b = (const float*)d_in[5];
    const float* cx2_1w = (const float*)d_in[6],  *cx2_1b = (const float*)d_in[7];
    const float* cx2_2w = (const float*)d_in[8],  *cx2_2b = (const float*)d_in[9];
    const float* cx3_1w = (const float*)d_in[10], *cx3_1b = (const float*)d_in[11];
    const float* cx3_2w = (const float*)d_in[12], *cx3_2b = (const float*)d_in[13];
    const float* cx4_1w = (const float*)d_in[14], *cx4_1b = (const float*)d_in[15];
    const float* cx4_2w = (const float*)d_in[16], *cx4_2b = (const float*)d_in[17];
    const float* l1w = (const float*)d_in[18], *l1b = (const float*)d_in[19];
    const float* l2w = (const float*)d_in[20], *l2b = (const float*)d_in[21];
    const float* l3w = (const float*)d_in[22], *l3b = (const float*)d_in[23];
    const float* l4w = (const float*)d_in[24], *l4b = (const float*)d_in[25];
    const float* l5w = (const float*)d_in[26], *l5b = (const float*)d_in[27];

    char* ws = (char*)d_ws;
    size_t off = 0;
    auto take = [&](size_t bytes) -> void* {
        void* p = ws + off;
        off += (bytes + 255) & ~(size_t)255;
        return p;
    };
    // Activation regions (aliased across phases; launches are stream-ordered):
    //   bufA: ctx hidden buffers -> concat1 [B,2176] -> concat3 [B,1152]
    //   bufB: concat2 [B,1152]   -> x4 [B,512]
    __bf16* stateBF = (__bf16*)take((size_t)BROWS * SDIM * 2);
    __bf16* bufA    = (__bf16*)take((size_t)BROWS * 2176 * 2);
    __bf16* bufB    = (__bf16*)take((size_t)BROWS * 1152 * 2);
    __bf16* wcx1 = (__bf16*)take((size_t)1024 * 1024 * 2);
    __bf16* wcx2 = (__bf16*)take((size_t)1024 * 1024 * 2);
    __bf16* wcx3 = (__bf16*)take((size_t)512 * 512 * 2);
    __bf16* w1   = (__bf16*)take((size_t)2048 * 128 * 2);
    __bf16* w2   = (__bf16*)take((size_t)1024 * 2176 * 2);
    __bf16* w3   = (__bf16*)take((size_t)1024 * 1152 * 2);
    __bf16* w4   = (__bf16*)take((size_t)512 * 1152 * 2);
    __bf16* w5   = (__bf16*)take((size_t)32 * 512 * 2);
    int* k1 = (int*)take(BROWS * 4);
    int* k2 = (int*)take(BROWS * 4);
    int* k3 = (int*)take(BROWS * 4);
    int* k4 = (int*)take(BROWS * 4);

    auto cvt = [&](const float* s, __bf16* d, long n) {
        cvt_bf16_kernel<<<(unsigned)((n + 255) / 256), 256, 0, stream>>>(s, d, n);
    };
    cvt(state,  stateBF, (long)BROWS * SDIM);
    cvt(cx1_2w, wcx1, 1024L * 1024);
    cvt(cx2_2w, wcx2, 1024L * 1024);
    cvt(cx3_2w, wcx3, 512L * 512);
    cvt(l1w, w1, 2048L * 128);
    cvt(l2w, w2, 1024L * 2176);
    cvt(l3w, w3, 1024L * 1152);
    cvt(l4w, w4, 512L * 1152);
    cvt(l5w, w5, 32L * 512);

    // Context nets -> per-row k values
    ctx4_kernel<<<BROWS / 256, 256, 0, stream>>>(task, cx4_1w, cx4_1b, cx4_2w, cx4_2b, k4);
    {
        long n = (long)BROWS * 1024;
        // smem = max(A stage 16*K*2, epilogue 16*N*4)
        ctx_hidden_kernel<<<(unsigned)((n + 255) / 256), 256, 0, stream>>>(task, cx1_1w, cx1_1b, bufA, 1024);
        gemm_argmax_kernel<1024, 1024, 8><<<BROWS / 16, 256, 16 * 1024 * 4, stream>>>(bufA, wcx1, cx1_2b, k1);
        ctx_hidden_kernel<<<(unsigned)((n + 255) / 256), 256, 0, stream>>>(task, cx2_1w, cx2_1b, bufA, 1024);
        gemm_argmax_kernel<1024, 1024, 8><<<BROWS / 16, 256, 16 * 1024 * 4, stream>>>(bufA, wcx2, cx2_2b, k2);
        long n3 = (long)BROWS * 512;
        ctx_hidden_kernel<<<(unsigned)((n3 + 255) / 256), 256, 0, stream>>>(task, cx3_1w, cx3_1b, bufA, 512);
        gemm_argmax_kernel<512, 512, 4><<<BROWS / 16, 256, 16 * 512 * 4, stream>>>(bufA, wcx3, cx3_2b, k3);
    }

    // Main chain (each layer: TDM A-stage + WMMA GEMM + bias + kwta + concat tail)
    gemm_kwta_kernel<2048, 128, 16, true><<<BROWS / 16, 256, 16 * 2048 * 4, stream>>>(
        stateBF, w1, l1b, k1, stateBF, bufA /*concat1*/, 2176, 1u);
    gemm_kwta_kernel<1024, 2176, 8, true><<<BROWS / 16, 256, 16 * 2176 * 2, stream>>>(
        bufA, w2, l2b, k2, stateBF, bufB /*concat2*/, 1152, 2u);
    gemm_kwta_kernel<1024, 1152, 8, true><<<BROWS / 16, 256, 16 * 1024 * 4, stream>>>(
        bufB, w3, l3b, k3, stateBF, bufA /*concat3*/, 1152, 3u);
    gemm_kwta_kernel<512, 1152, 4, false><<<BROWS / 16, 256, 16 * 1152 * 2, stream>>>(
        bufA, w4, l4b, k4, stateBF, bufB /*x4*/, 512, 4u);
    gemm_out_kernel<<<BROWS / 128, 256, 128 * 512 * 2, stream>>>(bufB, w5, l5b, (float*)d_out);
}
